// MHA_85426899517653
// MI455X (gfx1250) — matrix operand order repaired and checked by reading, not compiled, not run on hardware
//
#include <hip/hip_runtime.h>
#include <hip/hip_bf16.h>
#include <math.h>

// ---------------------------------------------------------------------------
// MHA for MI455X (gfx1250, wave32).  D_MODEL=1024, H=16, dh=64, B=4, L=2048.
// GEMMs on v_wmma_f32_16x16x32_bf16; B operand staged via async-to-LDS DMA
// (global_load_async_to_lds_b128, ASYNCcnt double-buffering), shared by 8
// waves per workgroup.  Steady-state K-loop is branchless (last step peeled)
// and kept at unroll 1 to avoid accumulator-copy churn in RA.
// ---------------------------------------------------------------------------

typedef __attribute__((ext_vector_type(16))) __bf16 v16bf;
typedef __attribute__((ext_vector_type(8)))  __bf16 v8bf;
typedef __attribute__((ext_vector_type(8)))  float  v8f;

#define D_MODEL 1024
#define N_HEADS 16
#define D_HEAD  64
#define BATCH   4
#define SEQ     2048
#define MTOT    (BATCH * SEQ)          // 8192 rows

// Build a v16bf A/B operand from two contiguous 8-element (16B) chunks.
__device__ __forceinline__ v16bf ld16(const __bf16* lo, const __bf16* hi) {
    v8bf a = *(const v8bf*)lo;
    v8bf b = *(const v8bf*)hi;
    v16bf r;
#pragma unroll
    for (int i = 0; i < 8; ++i) { r[i] = a[i]; r[i + 8] = b[i]; }
    return r;
}

__device__ __forceinline__ v8f wmma_bf16(v16bf a, v16bf b, v8f c) {
    return __builtin_amdgcn_wmma_f32_16x16x32_bf16(false, a, false, b,
                                                   (short)0, c, false, false);
}

// CDNA5 async DMA: 16B global -> LDS, tracked by ASYNCcnt.
__device__ __forceinline__ void async_ld16(unsigned lds_off, const void* g) {
    unsigned long long ga = (unsigned long long)(uintptr_t)g;
    asm volatile("global_load_async_to_lds_b128 %0, %1, off"
                 :: "v"(lds_off), "v"(ga) : "memory");
}

// ---------------------------------------------------------------------------
// fp32 -> bf16 grid-stride convert
// ---------------------------------------------------------------------------
__global__ void f32_to_bf16(const float* __restrict__ s, __bf16* __restrict__ d, int n) {
    for (int i = blockIdx.x * blockDim.x + threadIdx.x; i < n;
         i += gridDim.x * blockDim.x)
        d[i] = (__bf16)s[i];
}

// Transpose 1024x1024 fp32 weight [K,N] -> bf16 [N,K]
__global__ void w_transpose(const float* __restrict__ W, __bf16* __restrict__ Wt) {
    int i = blockIdx.x * blockDim.x + threadIdx.x;   // 0 .. 1048575
    int k = i >> 10, n = i & 1023;
    Wt[(size_t)n * D_MODEL + k] = (__bf16)W[i];
}

// ---------------------------------------------------------------------------
// Projection GEMM with async-LDS-staged B operand.
// Block = 256 threads (8 waves) computes a 256x64 tile; each wave 32x64.
// B slab (64 cols x 32 k, bf16, 4KB) double-buffered in LDS; slab for step
// k+32 is DMA'd (one global_load_async_to_lds_b128 per thread) while step k
// computes.  Last K-step peeled so the hot loop is branchless.
// mode 0: Q bf16 [B,H,L,dh]   mode 1: K bf16 [B,H,L,dh]
// mode 2: Vt bf16 [B,H,dh,L]  mode 3: fp32 [M, D_MODEL]
// ---------------------------------------------------------------------------
__global__ __launch_bounds__(256) void proj_gemm(const __bf16* __restrict__ A,
                                                 const __bf16* __restrict__ Wt,
                                                 const float* __restrict__ bias,
                                                 void* __restrict__ outp, int mode) {
    __shared__ __align__(16) __bf16 bslab[2][64 * 32];   // 2 x 4KB

    const int tid  = threadIdx.x;
    const int wv   = tid >> 5;
    const int lane = tid & 31;
    const int hf   = lane >> 4;
    const int ln   = lane & 15;

    const int mblk = (blockIdx.x >> 4) << 8;     // 256-row block
    const int n0   = (blockIdx.x & 15) << 6;     // 64-col block
    const int mw   = mblk + wv * 32;             // this wave's 32 rows

    const __bf16* arow0 = A + (size_t)(mw + ln) * D_MODEL;
    const __bf16* arow1 = A + (size_t)(mw + 16 + ln) * D_MODEL;

    // slab DMA: 256 threads x 16B = 4KB.  row = tid/4, chunk = tid&3.
    const int srow = tid >> 2;
    const int schk = (tid & 3) << 3;             // element offset 0/8/16/24
    const __bf16* sgbase = Wt + (size_t)(n0 + srow) * D_MODEL + schk;
    const unsigned slab0 = (unsigned)(uintptr_t)&bslab[0][0];
    const unsigned sloff = (unsigned)(srow * 64 + schk * 2);  // bytes in slab

    v8f acc[2][4];
#pragma unroll
    for (int ms = 0; ms < 2; ++ms)
#pragma unroll
        for (int t = 0; t < 4; ++t) acc[ms][t] = (v8f)0.0f;

    // one K-step of compute out of LDS buffer `cur`
    auto compute = [&](int k0, int cur) {
        // A operands (wave-private, direct global b128 loads)
        v16bf a0 = ld16(arow0 + k0 + 8 * hf, arow0 + k0 + 16 + 8 * hf);
        v16bf a1 = ld16(arow1 + k0 + 8 * hf, arow1 + k0 + 16 + 8 * hf);
        v16bf b[4];
#pragma unroll
        for (int t = 0; t < 4; ++t) {
            const __bf16* bp = &bslab[cur][(t * 16 + ln) * 32 + 8 * hf];
            b[t] = ld16(bp, bp + 16);
        }
#pragma unroll
        for (int t = 0; t < 4; ++t) {
            acc[0][t] = wmma_bf16(a0, b[t], acc[0][t]);
            acc[1][t] = wmma_bf16(a1, b[t], acc[1][t]);
        }
    };

    // prologue: DMA slab for k0 = 0 into buffer 0
    async_ld16(slab0 + sloff, sgbase);

    // steady state: branchless; prefetch k0+32 while computing k0.
    // unroll 1: x2 unrolling makes RA copy the 8 live acc tiles around.
#pragma unroll 1
    for (int k0 = 0; k0 < D_MODEL - 32; k0 += 32) {
        const int cur = (k0 >> 5) & 1;
        const unsigned nbase = slab0 + (unsigned)((cur ^ 1) * 64 * 32 * 2);
        async_ld16(nbase + sloff, sgbase + k0 + 32);
        asm volatile("s_wait_asynccnt 1" ::: "memory");
        __syncthreads();            // cur slab visible to all 8 waves
        compute(k0, cur);
        __syncthreads();            // all waves done before cur is overwritten
    }

    // peeled last step
    asm volatile("s_wait_asynccnt 0" ::: "memory");
    __syncthreads();
    compute(D_MODEL - 32, ((D_MODEL - 32) >> 5) & 1);

    const int h = n0 >> 6;           // head (mode 0/1/2)

#pragma unroll
    for (int ms = 0; ms < 2; ++ms) {
        const int m0   = mw + ms * 16;
        const int bidx = m0 >> 11;
        const int pos0 = (m0 & 2047) + 8 * hf;
#pragma unroll
        for (int t = 0; t < 4; ++t) {
            const int n  = n0 + t * 16 + ln;
            const float bv = bias[n];
            if (mode == 3) {
                float* O = (float*)outp;
#pragma unroll
                for (int r = 0; r < 8; ++r)
                    O[(size_t)(m0 + r + 8 * hf) * D_MODEL + n] = acc[ms][t][r] + bv;
            } else if (mode == 2) {
                __bf16* O = (__bf16*)outp;
                const int d = t * 16 + ln;
                v8bf pk;
#pragma unroll
                for (int r = 0; r < 8; ++r) pk[r] = (__bf16)(acc[ms][t][r] + bv);
                *(v8bf*)(O + ((size_t)(bidx * N_HEADS + h) * D_HEAD + d) * SEQ + pos0) = pk;
            } else {
                __bf16* O = (__bf16*)outp;
                const int d = t * 16 + ln;
#pragma unroll
                for (int r = 0; r < 8; ++r) {
                    const int pos = (m0 & 2047) + r + 8 * hf;
                    O[((size_t)(bidx * N_HEADS + h) * SEQ + pos) * D_HEAD + d] =
                        (__bf16)(acc[ms][t][r] + bv);
                }
            }
        }
    }
}

// ---------------------------------------------------------------------------
// Causal flash attention, 64-key blocks.  One wave per 16-query tile.
// Grid = 2048 blocks x 128 threads (4 waves).  ctx out: bf16 [M, D_MODEL].
// ---------------------------------------------------------------------------
__global__ __launch_bounds__(128) void attn_kernel(const __bf16* __restrict__ Q,
                                                   const __bf16* __restrict__ K,
                                                   const __bf16* __restrict__ Vt,
                                                   __bf16* __restrict__ ctx) {
    __shared__ __align__(16) __bf16 pbuf[4][16 * 64];  // per-wave P tile (2KB)

    const int wv   = threadIdx.x >> 5;
    const int lane = threadIdx.x & 31;
    const int hf   = lane >> 4;
    const int ln   = lane & 15;

    const int tIdx = blockIdx.x * 4 + wv;   // 0..8191
    const int qt   = tIdx & 127;
    const int h    = (tIdx >> 7) & 15;
    const int bidx = tIdx >> 11;
    const int q0   = qt * 16;

    const size_t bh = (size_t)(bidx * N_HEADS + h);
    const __bf16* qrow = Q + (bh * SEQ + q0 + ln) * D_HEAD;

    v16bf aQ[2];
#pragma unroll
    for (int ks = 0; ks < 2; ++ks)
        aQ[ks] = ld16(qrow + 32 * ks + 8 * hf, qrow + 32 * ks + 16 + 8 * hf);

    v8f acc[4];
#pragma unroll
    for (int t = 0; t < 4; ++t) acc[t] = (v8f)0.0f;
    float m_r[8], l_r[8];
#pragma unroll
    for (int r = 0; r < 8; ++r) { m_r[r] = -INFINITY; l_r[r] = 0.0f; }

    const float scale = 0.125f;              // 1/sqrt(64)
    const int nkb = (q0 + 16 + 63) >> 6;     // 64-key blocks (causal bound)

#pragma unroll 1
    for (int kb = 0; kb < nkb; ++kb) {
        const int k0 = kb * 64;

        // ---- scores: four 16x16 tiles over keys ----
        v8f s[4];
#pragma unroll
        for (int t = 0; t < 4; ++t) {
            s[t] = (v8f)0.0f;
            const __bf16* krow = K + (bh * SEQ + (k0 + 16 * t + ln)) * D_HEAD;
#pragma unroll
            for (int ks = 0; ks < 2; ++ks) {
                v16bf bK = ld16(krow + 32 * ks + 8 * hf,
                                krow + 32 * ks + 16 + 8 * hf);
                s[t] = wmma_bf16(aQ[ks], bK, s[t]);
            }
        }

        // ---- scale + causal mask + online softmax ----
#pragma unroll
        for (int r = 0; r < 8; ++r) {
            const int qi = q0 + r + 8 * hf;
            float sv[4];
#pragma unroll
            for (int t = 0; t < 4; ++t) {
                float v = s[t][r] * scale;
                if (k0 + 16 * t + ln > qi) v = -INFINITY;
                sv[t] = v;
            }
            float mx = fmaxf(fmaxf(sv[0], sv[1]), fmaxf(sv[2], sv[3]));
#pragma unroll
            for (int msk = 1; msk < 16; msk <<= 1)
                mx = fmaxf(mx, __shfl_xor(mx, msk, 32));

            const float m_new = fmaxf(m_r[r], mx);
            const float corr  = __expf(m_r[r] - m_new);
            float p[4], rs = 0.0f;
#pragma unroll
            for (int t = 0; t < 4; ++t) { p[t] = __expf(sv[t] - m_new); rs += p[t]; }
#pragma unroll
            for (int msk = 1; msk < 16; msk <<= 1)
                rs += __shfl_xor(rs, msk, 32);

            l_r[r] = l_r[r] * corr + rs;
            m_r[r] = m_new;
#pragma unroll
            for (int t = 0; t < 4; ++t) acc[t][r] *= corr;

#pragma unroll
            for (int t = 0; t < 4; ++t)
                pbuf[wv][(r + 8 * hf) * 64 + 16 * t + ln] = (__bf16)p[t];
        }

        __builtin_amdgcn_wave_barrier();

        // ---- P (A-layout) from LDS: two K-steps over the 64 keys ----
        const __bf16* pb = &pbuf[wv][ln * 64];
        v16bf aP[2];
#pragma unroll
        for (int ks = 0; ks < 2; ++ks)
            aP[ks] = ld16(pb + 32 * ks + 8 * hf, pb + 32 * ks + 16 + 8 * hf);

        __builtin_amdgcn_wave_barrier();

#pragma unroll
        for (int dt = 0; dt < 4; ++dt) {
            const __bf16* vrow = Vt + (bh * D_HEAD + dt * 16 + ln) * SEQ + k0;
#pragma unroll
            for (int ks = 0; ks < 2; ++ks) {
                v16bf bV = ld16(vrow + 32 * ks + 8 * hf,
                                vrow + 32 * ks + 16 + 8 * hf);
                acc[dt] = wmma_bf16(aP[ks], bV, acc[dt]);
            }
        }
    }

    // ---- normalize and store ctx bf16 [M, D_MODEL] ----
#pragma unroll
    for (int r = 0; r < 8; ++r) {
        const float inv = 1.0f / l_r[r];
        const size_t row = (size_t)(bidx * SEQ + q0 + r + 8 * hf);
#pragma unroll
        for (int dt = 0; dt < 4; ++dt)
            ctx[row * D_MODEL + h * D_HEAD + dt * 16 + ln] =
                (__bf16)(acc[dt][r] * inv);
    }
}

// ---------------------------------------------------------------------------
// Launch
// ---------------------------------------------------------------------------
extern "C" void kernel_launch(void* const* d_in, const int* in_sizes, int n_in,
                              void* d_out, int out_size, void* d_ws, size_t ws_size,
                              hipStream_t stream) {
    const float* x  = (const float*)d_in[0];
    // d_in[1] = attn_mask (tril) -> causal mask applied analytically
    const float* Wq = (const float*)d_in[2];
    const float* bq = (const float*)d_in[3];
    const float* Wk = (const float*)d_in[4];
    const float* bk = (const float*)d_in[5];
    const float* Wv = (const float*)d_in[6];
    const float* bv = (const float*)d_in[7];
    const float* Wo = (const float*)d_in[8];
    const float* bo = (const float*)d_in[9];

    char* ws = (char*)d_ws;
    const size_t SZ_X = (size_t)MTOT * D_MODEL * 2;      // 16 MB bf16
    const size_t SZ_W = (size_t)D_MODEL * D_MODEL * 2;   // 2 MB bf16
    __bf16* xb   = (__bf16*)(ws);
    __bf16* ctxb = (__bf16*)(ws + SZ_X);
    __bf16* Wqt  = (__bf16*)(ws + 2 * SZ_X);
    __bf16* Wkt  = (__bf16*)(ws + 2 * SZ_X + SZ_W);
    __bf16* Wvt  = (__bf16*)(ws + 2 * SZ_X + 2 * SZ_W);
    __bf16* Wot  = (__bf16*)(ws + 2 * SZ_X + 3 * SZ_W);
    __bf16* Qb   = (__bf16*)(ws + 2 * SZ_X + 4 * SZ_W);
    __bf16* Kb   = (__bf16*)(ws + 3 * SZ_X + 4 * SZ_W);
    __bf16* Vtb  = (__bf16*)(ws + 4 * SZ_X + 4 * SZ_W);

    // 1) conversions
    f32_to_bf16<<<4096, 256, 0, stream>>>(x, xb, MTOT * D_MODEL);
    w_transpose<<<4096, 256, 0, stream>>>(Wq, Wqt);
    w_transpose<<<4096, 256, 0, stream>>>(Wk, Wkt);
    w_transpose<<<4096, 256, 0, stream>>>(Wv, Wvt);
    w_transpose<<<4096, 256, 0, stream>>>(Wo, Wot);

    // 2) QKV projections: 32 m-blocks x 16 n-blocks = 512 blocks x 256 thr
    proj_gemm<<<512, 256, 0, stream>>>(xb, Wqt, bq, (void*)Qb, 0);
    proj_gemm<<<512, 256, 0, stream>>>(xb, Wkt, bk, (void*)Kb, 1);
    proj_gemm<<<512, 256, 0, stream>>>(xb, Wvt, bv, (void*)Vtb, 2);

    // 3) causal attention (8192 query tiles, 4 waves/block)
    attn_kernel<<<2048, 128, 0, stream>>>(Qb, Kb, Vtb, ctxb);

    // 4) output projection -> fp32 d_out
    proj_gemm<<<512, 256, 0, stream>>>(ctxb, Wot, bo, d_out, 3);
}
